// SelfAttention_68607807586747
// MI455X (gfx1250) — compile-verified, matrix operand order checked by reading, not run on hardware
//
#include <hip/hip_runtime.h>
#include <math.h>

typedef __attribute__((ext_vector_type(2)))  float  v2f;
typedef __attribute__((ext_vector_type(8)))  float  v8f;
typedef __attribute__((ext_vector_type(8)))  __bf16 v8bf;
typedef __attribute__((ext_vector_type(16))) __bf16 v16bf;

constexpr int B_ = 8, T_ = 1024, E_ = 128, H_ = 8;
constexpr int SROW = 1028;   // fp32 score row stride (1028 % 64 == 4 -> conflict-free)
constexpr int PROW = 1032;   // bf16 prob row stride (16B aligned; 516 dwords % 64 == 4)
constexpr float SCALE = 0.08838834764831845f;  // 1/sqrt(128)

static __device__ __forceinline__ v8f vzero8() {
  v8f z = {0.f, 0.f, 0.f, 0.f, 0.f, 0.f, 0.f, 0.f};
  return z;
}

// --- bf16 helpers (bit-exact, storage-only use of __bf16) ---
static __device__ __forceinline__ float bf2f(__bf16 h) {
  return __builtin_bit_cast(float, (unsigned)((unsigned)__builtin_bit_cast(unsigned short, h) << 16));
}
static __device__ __forceinline__ __bf16 f2bf(float f) {          // RNE
  unsigned u = __builtin_bit_cast(unsigned, f);
  u = (u + 0x7FFFu + ((u >> 16) & 1u)) >> 16;
  return __builtin_bit_cast(__bf16, (unsigned short)u);
}
static __device__ __forceinline__ void split2(float f, __bf16& h, __bf16& l) {
  h = f2bf(f);
  l = f2bf(f - bf2f(h));
}

// D = A(16x32 bf16) * B(32x16 bf16) + C(16x16 f32)
static __device__ __forceinline__ v8f wmma_bf(v16bf a, v16bf b, v8f c) {
  return __builtin_amdgcn_wmma_f32_16x16x32_bf16(false, a, false, b, (short)0, c, false, false);
}

// Per-lane A/B operand slice from a K-contiguous row.
// p must be pre-offset by row*stride + 8*(lane>>4).
// elems 0..7 <- K[k0 + 8*lh .. +7], elems 8..15 <- K[k0 + 16 + 8*lh .. +7]
static __device__ __forceinline__ v16bf ldAB(const __bf16* p, int k0) {
  v8bf a = *(const v8bf*)(p + k0);
  v8bf b = *(const v8bf*)(p + k0 + 16);
  return __builtin_shufflevector(a, b, 0,1,2,3,4,5,6,7,8,9,10,11,12,13,14,15);
}

// One K=32 chunk of the split-bf16 product: hi*hi + hi*lo + lo*hi
#define SPLIT_CHUNK(AH_, AL_, BH_, BL_, K0_, ACC_)  \
  do {                                              \
    v16bf _ah = ldAB((AH_), (K0_));                 \
    v16bf _al = ldAB((AL_), (K0_));                 \
    v16bf _bh = ldAB((BH_), (K0_));                 \
    v16bf _bl = ldAB((BL_), (K0_));                 \
    (ACC_) = wmma_bf(_ah, _bh, (ACC_));             \
    (ACC_) = wmma_bf(_ah, _bl, (ACC_));             \
    (ACC_) = wmma_bf(_al, _bh, (ACC_));             \
  } while (0)

// Same chunk but with the A operand (hi,lo) already resident in registers.
#define QK_CHUNK(QH_, QL_, BH_, BL_, K0_, ACC_)     \
  do {                                              \
    v16bf _bh = ldAB((BH_), (K0_));                 \
    v16bf _bl = ldAB((BL_), (K0_));                 \
    (ACC_) = wmma_bf((QH_), _bh, (ACC_));           \
    (ACC_) = wmma_bf((QH_), _bl, (ACC_));           \
    (ACC_) = wmma_bf((QL_), _bh, (ACC_));           \
  } while (0)

// ---------------------------------------------------------------------------
// Kernel 0: elementwise fp32 -> (bf16 hi, bf16 lo) split
// ---------------------------------------------------------------------------
__global__ __launch_bounds__(256)
void split_kernel(const float* __restrict__ src, __bf16* __restrict__ h,
                  __bf16* __restrict__ l, int n) {
  for (int i = blockIdx.x * blockDim.x + threadIdx.x; i < n; i += gridDim.x * blockDim.x) {
    const float f = src[i];
    __bf16 hh, ll;
    split2(f, hh, ll);
    h[i] = hh;
    l[i] = ll;
  }
}

// ---------------------------------------------------------------------------
// Kernel 1: QKV projection (split-bf16 GEMM, K=128 -> 4 chunks * 3 WMMA).
// grid = (512, H, 3), block = 256.  z==2 (V) stores transposed (B,H,E,T).
// ---------------------------------------------------------------------------
__global__ __launch_bounds__(256)
void qkv_kernel(const __bf16* __restrict__ xh, const __bf16* __restrict__ xl,
                const __bf16* __restrict__ wqh, const __bf16* __restrict__ wql,
                const __bf16* __restrict__ wkh, const __bf16* __restrict__ wkl,
                const __bf16* __restrict__ wvh, const __bf16* __restrict__ wvl,
                __bf16* __restrict__ qh, __bf16* __restrict__ ql,
                __bf16* __restrict__ kh, __bf16* __restrict__ kl,
                __bf16* __restrict__ vth, __bf16* __restrict__ vtl) {
  const int wave = threadIdx.x >> 5, lane = threadIdx.x & 31;
  const int r16 = lane & 15, lh = lane >> 4;

  const __bf16 *Wh, *Wl; __bf16 *oh, *ol; bool transposed = false;
  if (blockIdx.z == 0)      { Wh = wqh; Wl = wql; oh = qh;  ol = ql; }
  else if (blockIdx.z == 1) { Wh = wkh; Wl = wkl; oh = kh;  ol = kl; }
  else                      { Wh = wvh; Wl = wvl; oh = vth; ol = vtl; transposed = true; }

  const int m0 = blockIdx.x * 16;
  const int h  = blockIdx.y;
  const int n0 = h * E_ + wave * 16;

  const __bf16* axh = xh + (size_t)(m0 + r16) * E_ + 8 * lh;
  const __bf16* axl = xl + (size_t)(m0 + r16) * E_ + 8 * lh;
  const __bf16* bwh = Wh + (size_t)(n0 + r16) * E_ + 8 * lh;
  const __bf16* bwl = Wl + (size_t)(n0 + r16) * E_ + 8 * lh;

  v8f acc0 = vzero8(), acc1 = vzero8(), acc2 = vzero8(), acc3 = vzero8();
  SPLIT_CHUNK(axh, axl, bwh, bwl, 0,  acc0);
  SPLIT_CHUNK(axh, axl, bwh, bwl, 32, acc1);
  SPLIT_CHUNK(axh, axl, bwh, bwl, 64, acc2);
  SPLIT_CHUNK(axh, axl, bwh, bwl, 96, acc3);
  v8f acc = (acc0 + acc1) + (acc2 + acc3);

  const int bb = m0 / T_;
  const int tb = m0 % T_;
  const int e  = (n0 & (E_ - 1)) + r16;
#pragma unroll
  for (int j = 0; j < 8; ++j) {
    const int t = tb + j + 8 * lh;               // C layout: VGPR j -> M = j + 8*(lane/16)
    __bf16 hh, ll;
    split2(acc[j], hh, ll);
    size_t idx;
    if (transposed) idx = (((size_t)bb * H_ + h) * E_ + e) * T_ + t;   // (B,H,E,T)
    else            idx = (((size_t)bb * H_ + h) * T_ + t) * E_ + e;   // (B,H,T,E)
    oh[idx] = hh;
    ol[idx] = ll;
  }
}

// ---------------------------------------------------------------------------
// Kernel 2: attention, one (b,h) x 32 query rows per block.
// LDS: fp32 S [32][SROW] + bf16 Phi/Plo [32][PROW] = 263,680 B.
// ---------------------------------------------------------------------------
__global__ __launch_bounds__(256)
void attn_kernel(const __bf16* __restrict__ qh, const __bf16* __restrict__ ql,
                 const __bf16* __restrict__ kh, const __bf16* __restrict__ kl,
                 const __bf16* __restrict__ vth, const __bf16* __restrict__ vtl,
                 const int* __restrict__ mask,
                 __bf16* __restrict__ aoh, __bf16* __restrict__ aol) {
  extern __shared__ char smem[];
  float*  sS  = (float*)smem;                          // [32][SROW] fp32
  __bf16* pHi = (__bf16*)(smem + 32 * SROW * 4);       // [32][PROW]
  __bf16* pLo = pHi + 32 * PROW;                       // [32][PROW]

  const int wave = threadIdx.x >> 5, lane = threadIdx.x & 31;
  const int r16 = lane & 15, lh = lane >> 4;
  const int b = blockIdx.z, h = blockIdx.y, t0 = blockIdx.x * 32;

  const size_t bh = ((size_t)b * H_ + h) * (size_t)T_ * E_;
  const __bf16* Qh = qh + bh;  const __bf16* Ql = ql + bh;
  const __bf16* Kh = kh + bh;  const __bf16* Kl = kl + bh;
  const __bf16* Vh = vth + bh; const __bf16* Vl = vtl + bh;   // (E,T) layout

  // ---- Phase 1: S = (Q K^T)*scale, causal -inf then padding -1e9 ----
  // Q operand (8 x v16bf) is hoisted into registers and reused for all 8
  // key-tiles this wave handles at a given tm -> ~2x fewer VMEM issues.
  for (int tm = 0; tm < 2; ++tm) {
    const int q0 = t0 + tm * 16;
    const __bf16* aqh = Qh + (size_t)(q0 + r16) * E_ + 8 * lh;
    const __bf16* aql = Ql + (size_t)(q0 + r16) * E_ + 8 * lh;
    const v16bf q_h0 = ldAB(aqh, 0),  q_h1 = ldAB(aqh, 32);
    const v16bf q_h2 = ldAB(aqh, 64), q_h3 = ldAB(aqh, 96);
    const v16bf q_l0 = ldAB(aql, 0),  q_l1 = ldAB(aql, 32);
    const v16bf q_l2 = ldAB(aql, 64), q_l3 = ldAB(aql, 96);

    for (int tn = wave; tn < 64; tn += 8) {
      const int k0 = tn * 16;
      const __bf16* bkh = Kh + (size_t)(k0 + r16) * E_ + 8 * lh;
      const __bf16* bkl = Kl + (size_t)(k0 + r16) * E_ + 8 * lh;

      v8f acc0 = vzero8(), acc1 = vzero8(), acc2 = vzero8(), acc3 = vzero8();
      QK_CHUNK(q_h0, q_l0, bkh, bkl, 0,  acc0);
      QK_CHUNK(q_h1, q_l1, bkh, bkl, 32, acc1);
      QK_CHUNK(q_h2, q_l2, bkh, bkl, 64, acc2);
      QK_CHUNK(q_h3, q_l3, bkh, bkl, 96, acc3);
      v8f acc = (acc0 + acc1) + (acc2 + acc3);

      const int kg = k0 + r16;
#pragma unroll
      for (int j = 0; j < 8; ++j) {
        const int qg = q0 + j + 8 * lh;
        float val = acc[j] * SCALE;
        if (kg > qg) val = -INFINITY;                      // causal first
        if (mask[(size_t)qg * T_ + kg] == 0) val = -1e9f;  // padding overwrites
        sS[(tm * 16 + j + 8 * lh) * SROW + kg] = val;
      }
    }
  }
  __syncthreads();

  // ---- Phase 2: row softmax; emit P as split bf16 ----
  for (int rr = wave * 4; rr < wave * 4 + 4; ++rr) {
    float* row = sS + rr * SROW;
    float mx = -INFINITY;
    for (int c = lane; c < T_; c += 32) mx = fmaxf(mx, row[c]);
    for (int off = 16; off; off >>= 1) mx = fmaxf(mx, __shfl_xor(mx, off, 32));
    float sum = 0.f;
    for (int c = lane; c < T_; c += 32) { float e = expf(row[c] - mx); row[c] = e; sum += e; }
    for (int off = 16; off; off >>= 1) sum += __shfl_xor(sum, off, 32);
    const float inv = 1.f / sum;
    for (int c = lane; c < T_; c += 32) {
      __bf16 hh, ll;
      split2(row[c] * inv, hh, ll);
      pHi[rr * PROW + c] = hh;
      pLo[rr * PROW + c] = ll;
    }
  }
  __syncthreads();

  // ---- Phase 3: O = P @ V  (A from LDS split-bf16, B from transposed V) ----
  for (int tix = wave; tix < 16; tix += 8) {        // 2 x 8 tiles of 16x16
    const int tm = tix >> 3, tn = tix & 7;
    const __bf16* aph = pHi + (tm * 16 + r16) * PROW + 8 * lh;
    const __bf16* apl = pLo + (tm * 16 + r16) * PROW + 8 * lh;
    const __bf16* bvh = Vh + (size_t)(tn * 16 + r16) * T_ + 8 * lh;
    const __bf16* bvl = Vl + (size_t)(tn * 16 + r16) * T_ + 8 * lh;

    v8f acc0 = vzero8(), acc1 = vzero8(), acc2 = vzero8(), acc3 = vzero8();
#pragma unroll 2
    for (int k0 = 0; k0 < T_; k0 += 128) {
      SPLIT_CHUNK(aph, apl, bvh, bvl, k0,      acc0);
      SPLIT_CHUNK(aph, apl, bvh, bvl, k0 + 32, acc1);
      SPLIT_CHUNK(aph, apl, bvh, bvl, k0 + 64, acc2);
      SPLIT_CHUNK(aph, apl, bvh, bvl, k0 + 96, acc3);
    }
    v8f acc = (acc0 + acc1) + (acc2 + acc3);

#pragma unroll
    for (int j = 0; j < 8; ++j) {
      const int t = t0 + tm * 16 + j + 8 * lh;
      const size_t idx = ((size_t)b * T_ + t) * (H_ * E_) + h * E_ + tn * 16 + r16;
      __bf16 hh, ll;
      split2(acc[j], hh, ll);
      aoh[idx] = hh;       // (B, T, H*E) pre-split for the output projection
      aol[idx] = ll;
    }
  }
}

// ---------------------------------------------------------------------------
// Kernel 3: out = ao @ Wu^T + bu   (split-bf16, M=8192, N=128, K=1024)
// ---------------------------------------------------------------------------
__global__ __launch_bounds__(256)
void proj_kernel(const __bf16* __restrict__ aoh, const __bf16* __restrict__ aol,
                 const __bf16* __restrict__ wuh, const __bf16* __restrict__ wul,
                 const float* __restrict__ bu, float* __restrict__ out) {
  const int wave = threadIdx.x >> 5, lane = threadIdx.x & 31;
  const int r16 = lane & 15, lh = lane >> 4;
  const int m0 = blockIdx.x * 16;
  const int n0 = wave * 16;
  const int K  = H_ * E_;  // 1024

  const __bf16* aah = aoh + (size_t)(m0 + r16) * K + 8 * lh;
  const __bf16* aal = aol + (size_t)(m0 + r16) * K + 8 * lh;
  const __bf16* bwh = wuh + (size_t)(n0 + r16) * K + 8 * lh;
  const __bf16* bwl = wul + (size_t)(n0 + r16) * K + 8 * lh;

  v8f acc0 = vzero8(), acc1 = vzero8(), acc2 = vzero8(), acc3 = vzero8();
#pragma unroll 2
  for (int k0 = 0; k0 < K; k0 += 128) {
    SPLIT_CHUNK(aah, aal, bwh, bwl, k0,      acc0);
    SPLIT_CHUNK(aah, aal, bwh, bwl, k0 + 32, acc1);
    SPLIT_CHUNK(aah, aal, bwh, bwl, k0 + 64, acc2);
    SPLIT_CHUNK(aah, aal, bwh, bwl, k0 + 96, acc3);
  }
  v8f acc = (acc0 + acc1) + (acc2 + acc3);

  const float bias = bu[n0 + r16];
#pragma unroll
  for (int j = 0; j < 8; ++j) {
    out[(size_t)(m0 + j + 8 * lh) * E_ + n0 + r16] = acc[j] + bias;
  }
}

// ---------------------------------------------------------------------------
extern "C" void kernel_launch(void* const* d_in, const int* in_sizes, int n_in,
                              void* d_out, int out_size, void* d_ws, size_t ws_size,
                              hipStream_t stream) {
  (void)in_sizes; (void)n_in; (void)out_size; (void)ws_size;
  const float* x    = (const float*)d_in[0];
  const int*   mask = (const int*)  d_in[1];
  const float* Wk   = (const float*)d_in[2];   // dict order: x, mask, Wk, Wq, Wv, Wu, bu
  const float* Wq   = (const float*)d_in[3];
  const float* Wv   = (const float*)d_in[4];
  const float* Wu   = (const float*)d_in[5];
  const float* bu   = (const float*)d_in[6];
  float* out = (float*)d_out;

  // workspace layout (bf16 element offsets); total ~140.5 MB
  __bf16* ws = (__bf16*)d_ws;
  const size_t CH = (size_t)B_ * H_ * T_ * E_;   // 8,388,608
  const size_t XH = (size_t)B_ * T_ * E_;        // 1,048,576
  const size_t WH = (size_t)(H_ * E_) * E_;      //   131,072
  __bf16* qh  = ws;            __bf16* ql  = ws + CH;
  __bf16* kh  = ws + 2 * CH;   __bf16* kl  = ws + 3 * CH;
  __bf16* vth = ws + 4 * CH;   __bf16* vtl = ws + 5 * CH;
  __bf16* aoh = ws + 6 * CH;   __bf16* aol = ws + 7 * CH;
  __bf16* xh  = ws + 8 * CH;   __bf16* xl  = xh + XH;
  __bf16* wb  = xl + XH;
  __bf16* wqh = wb;            __bf16* wql = wb + WH;
  __bf16* wkh = wb + 2 * WH;   __bf16* wkl = wb + 3 * WH;
  __bf16* wvh = wb + 4 * WH;   __bf16* wvl = wb + 5 * WH;
  __bf16* wuh = wb + 6 * WH;   __bf16* wul = wb + 7 * WH;

  // 0) split fp32 inputs into bf16 hi/lo
  split_kernel<<<512, 256, 0, stream>>>(x,  xh,  xl,  (int)XH);
  split_kernel<<<128, 256, 0, stream>>>(Wq, wqh, wql, (int)WH);
  split_kernel<<<128, 256, 0, stream>>>(Wk, wkh, wkl, (int)WH);
  split_kernel<<<128, 256, 0, stream>>>(Wv, wvh, wvl, (int)WH);
  split_kernel<<<128, 256, 0, stream>>>(Wu, wuh, wul, (int)WH);

  // 1) QKV projection (V transposed)
  dim3 g1((B_ * T_) / 16, H_, 3);
  qkv_kernel<<<g1, 256, 0, stream>>>(xh, xl, wqh, wql, wkh, wkl, wvh, wvl,
                                     qh, ql, kh, kl, vth, vtl);

  // 2) attention
  const size_t lds = (size_t)32 * SROW * 4 + (size_t)2 * 32 * PROW * 2;  // 263,680 B
  (void)hipFuncSetAttribute((const void*)attn_kernel,
                            hipFuncAttributeMaxDynamicSharedMemorySize, (int)lds);
  dim3 g2(T_ / 32, H_, B_);
  attn_kernel<<<g2, 256, lds, stream>>>(qh, ql, kh, kl, vth, vtl, mask, aoh, aol);

  // 3) output projection
  proj_kernel<<<(B_ * T_) / 16, 256, 0, stream>>>(aoh, aol, wuh, wul, bu, out);
}
